// MultiHeadedAttention_50594714747320
// MI455X (gfx1250) — matrix operand order repaired and checked by reading, not compiled, not run on hardware
//
#include <hip/hip_runtime.h>
#include <hip/hip_bf16.h>

// ---------------------------------------------------------------------------
// MHA for MI455X (gfx1250): bf16 WMMA (v_wmma_f32_16x16x32_bf16), wave32.
// Pipeline: f32->bf16 convert -> Q/K/V projections (zero-LDS WMMA GEMM)
//           -> per-(b,h,qtile16) chunked flash attention (512-key chunks in
//              33KB LDS, online softmax rescale, P@V WMMA)
//           -> output projection (WMMA GEMM, fp32 out).
// Chunking keeps LDS small so ~8 workgroups/WGP co-reside (vs 2 with a full
// 16x2048 fp32 score buffer), hiding global-load and v_exp latency.
// ---------------------------------------------------------------------------

#define HH   16
#define DM   1024
#define DK   64
#define SS   2048
#define BB   2
#define MROWS (BB * SS)   // 4096
#define CH   512          // keys per attention chunk
#define SST  516          // LDS row stride in floats (bank de-phasing)

typedef __bf16 bf16_t;
typedef bf16_t v16bf __attribute__((ext_vector_type(16)));
typedef bf16_t v8bf  __attribute__((ext_vector_type(8)));
typedef float  v8f   __attribute__((ext_vector_type(8)));

__device__ __forceinline__ v8f wmma_bf16(v16bf a, v16bf b, v8f c) {
  // (neg_a, A, neg_b, B, c_mod, C, reuse_a, reuse_b)
  return __builtin_amdgcn_wmma_f32_16x16x32_bf16(false, a, false, b,
                                                 (short)0, c, false, false);
}

__device__ __forceinline__ v16bf cat8(v8bf lo, v8bf hi) {
  return __builtin_shufflevector(lo, hi, 0, 1, 2, 3, 4, 5, 6, 7,
                                         8, 9, 10, 11, 12, 13, 14, 15);
}

__device__ __forceinline__ v8f zero8() {
  v8f z;
#pragma unroll
  for (int i = 0; i < 8; ++i) z[i] = 0.0f;
  return z;
}

// ------------------------------ f32 -> bf16 --------------------------------
__global__ void cvt_kernel(const float* __restrict__ src,
                           bf16_t* __restrict__ dst, int n) {
  int i = blockIdx.x * blockDim.x + threadIdx.x;
  int stride = gridDim.x * blockDim.x;
  for (; i < n; i += stride) dst[i] = (bf16_t)src[i];
}

// ------------------------- zero-LDS WMMA GEMM ------------------------------
// Y[M=4096, N=1024] = X[M,K=1024] @ W[N,K]^T + bias.
// One wave owns a 16x64 tile. A: 2 x b128/lane from X row. B: 1 x 32B/lane
// from W row (column n of W^T == row n of W, contiguous in K).
// MODE 0: write bf16 head-split  [B,H,S,64]   (Q, K)
// MODE 1: write bf16 transposed  [B,H,64,S]   (V^T)
// MODE 2: write fp32 plain       [M,N]        (final output)
template <int MODE>
__global__ __launch_bounds__(128) void gemm_kernel(
    const bf16_t* __restrict__ X, const bf16_t* __restrict__ W,
    const float* __restrict__ bias, void* __restrict__ dstv) {
  const int lane = threadIdx.x & 31;
  const int wave = threadIdx.x >> 5;
  const int tile = blockIdx.x * 4 + wave;      // 4096 tiles total
  const int NT = DM / 64;                      // 16 n-tiles
  const int mt = tile / NT;
  const int nt = tile % NT;
  const int ln  = lane & 15;
  const int sel = lane >> 4;

  const bf16_t* xrow = X + (size_t)(mt * 16 + ln) * DM;

  v8f acc[4];
#pragma unroll
  for (int c = 0; c < 4; ++c) acc[c] = zero8();

  for (int kk = 0; kk < DM / 32; ++kk) {
    const int kb = kk * 32;
    // A fragment 16x32: lane row = ln; sel=0 -> K 0..15, sel=1 -> K 16..31
    v8bf alo = *(const v8bf*)(xrow + kb + sel * 16);
    v8bf ahi = *(const v8bf*)(xrow + kb + sel * 16 + 8);
    v16bf a = cat8(alo, ahi);
#pragma unroll
    for (int c = 0; c < 4; ++c) {
      // B fragment 32x16: lane column = nt*64 + c*16 + ln;
      // lanes 0-15 hold K kb+0..15, lanes 16-31 hold K kb+16..31.
      const bf16_t* wrow =
          W + (size_t)(nt * 64 + c * 16 + ln) * DM + kb + sel * 16;
      v16bf bfr = *(const v16bf*)wrow;
      acc[c] = wmma_bf16(a, bfr, acc[c]);
    }
  }

#pragma unroll
  for (int c = 0; c < 4; ++c) {
    const int gcol = nt * 64 + c * 16 + ln;
    const float bv = bias[gcol];
#pragma unroll
    for (int r = 0; r < 8; ++r) {
      const int grow = mt * 16 + r + 8 * sel;  // C/D: lanes16-31 hold M=r+8
      const float val = acc[c][r] + bv;
      if (MODE == 2) {
        ((float*)dstv)[(size_t)grow * DM + gcol] = val;
      } else {
        const int b = grow >> 11, s = grow & (SS - 1);
        const int h = gcol >> 6,  d = gcol & 63;
        bf16_t* dst = (bf16_t*)dstv;
        if (MODE == 0)
          dst[(size_t)((b * HH + h) * SS + s) * DK + d] = (bf16_t)val;
        else
          dst[(size_t)((b * HH + h) * DK + d) * SS + s] = (bf16_t)val;
      }
    }
  }
}

// --------------------- chunked flash attention -----------------------------
// One wave per (b, h, 16-row q tile).  Keys processed in CH=512 chunks.
// LDS: 16 x SST fp32 chunk scores + rmax/rsum/factor[16]  (~33 KB).
__global__ __launch_bounds__(32) void attn_kernel(
    const bf16_t* __restrict__ Qb, const bf16_t* __restrict__ Kb,
    const bf16_t* __restrict__ Vt, const int* __restrict__ mask,
    bf16_t* __restrict__ ctx) {
  extern __shared__ float smem[];
  float* sc    = smem;                   // [16][SST]
  float* rmaxv = smem + 16 * SST;        // [16] running row max
  float* rsumv = rmaxv + 16;             // [16] running row sum
  float* facv  = rsumv + 16;             // [16] rescale factor this chunk

  const int lane = threadIdx.x;
  const int ln  = lane & 15;
  const int sel = lane >> 4;
  const int bid = blockIdx.x;
  const int qt = bid & (SS / 16 - 1);    // 128 q-tiles
  const int bh = bid >> 7;               // b*H + h
  const int b  = bh >> 4;
  const int h  = bh & 15;
  const int qbase = qt * 16;

  if (lane < 16) {
    rmaxv[lane] = -3.0e38f;
    rsumv[lane] = 0.0f;
  }

  // Q A-fragments for d_k chunks 0..31 and 32..63 (kept in VGPRs throughout)
  const bf16_t* qrow = Qb + (size_t)(bh * SS + qbase + ln) * DK;
  v16bf aq[2];
#pragma unroll
  for (int j = 0; j < 2; ++j) {
    v8bf lo = *(const v8bf*)(qrow + j * 32 + sel * 16);
    v8bf hi = *(const v8bf*)(qrow + j * 32 + sel * 16 + 8);
    aq[j] = cat8(lo, hi);
  }

  const bf16_t* kmat  = Kb + (size_t)bh * SS * DK;
  const bf16_t* vbase = Vt + (size_t)bh * DK * SS;
  const int*    mrow0 = mask + (size_t)(b * SS + qbase + ln) * SS;

  v8f o[4];
#pragma unroll
  for (int c = 0; c < 4; ++c) o[c] = zero8();

  __syncthreads();

  for (int kc = 0; kc < SS / CH; ++kc) {
    const int keybase = kc * CH;

    // ---- chunk scores = Q @ K^T (raw f32 into LDS) ----
    for (int kt = 0; kt < CH / 16; ++kt) {
      const bf16_t* krow = kmat + (size_t)(keybase + kt * 16 + ln) * DK;
      __builtin_prefetch(krow + 16 * DK, 0, 3);  // next tile's K row -> L2/L0
      v8f acc = zero8();
#pragma unroll
      for (int j = 0; j < 2; ++j) {
        v16bf bk = *(const v16bf*)(krow + j * 32 + sel * 16);
        acc = wmma_bf16(aq[j], bk, acc);
      }
#pragma unroll
      for (int r = 0; r < 8; ++r)
        sc[(r + 8 * sel) * SST + kt * 16 + ln] = acc[r];
    }
    __syncthreads();

    // ---- online softmax over the chunk (two lanes per row) ----
    {
      float* rowp = sc + ln * SST;
      const int* mrow = mrow0 + keybase;
      const int c0 = sel * (CH / 2);
      float mx = -3.0e38f;
      for (int c = 0; c < CH / 2; ++c) {
        const int col = c0 + c;
        float v = rowp[col] * 0.125f;      // 1/sqrt(64)
        if (mrow[col] == 0) v = -1.0e9f;
        rowp[col] = v;
        mx = fmaxf(mx, v);
      }
      mx = fmaxf(mx, __shfl_xor(mx, 16));
      const float oldmax = rmaxv[ln];
      const float newmax = fmaxf(oldmax, mx);
      const float fct = __expf(oldmax - newmax);  // 0 on first chunk
      float sum = 0.0f;
      for (int c = 0; c < CH / 2; ++c) {
        const int col = c0 + c;
        const float e = __expf(rowp[col] - newmax);
        rowp[col] = e;
        sum += e;
      }
      sum += __shfl_xor(sum, 16);
      if (sel == 0) {
        rmaxv[ln] = newmax;
        rsumv[ln] = rsumv[ln] * fct + sum;
        facv[ln]  = fct;
      }
    }
    __syncthreads();

    // ---- rescale running output accumulators ----
#pragma unroll
    for (int c = 0; c < 4; ++c) {
#pragma unroll
      for (int r = 0; r < 8; ++r) o[c][r] *= facv[r + 8 * sel];
    }

    // ---- O += P @ V over this chunk (V stored transposed: [d, s]) ----
    for (int kt2 = 0; kt2 < CH / 32; ++kt2) {
      const float* p0 = sc + ln * SST + kt2 * 32 + sel * 16;
      const float* p1 = p0 + 8;
      v8bf lo, hi;
#pragma unroll
      for (int i = 0; i < 8; ++i) {
        lo[i] = (bf16_t)p0[i];
        hi[i] = (bf16_t)p1[i];
      }
      v16bf ap = cat8(lo, hi);
#pragma unroll
      for (int c = 0; c < 4; ++c) {
        const bf16_t* vrow =
            vbase + (size_t)(c * 16 + ln) * SS + keybase + kt2 * 32 + sel * 16;
        v16bf bv = *(const v16bf*)vrow;
        o[c] = wmma_bf16(ap, bv, o[c]);
      }
    }
    __syncthreads();   // chunk scores consumed before next chunk overwrites
  }

  // ---- normalize by final rowsum, write merged-head ctx [4096,1024] ----
#pragma unroll
  for (int c = 0; c < 4; ++c) {
#pragma unroll
    for (int r = 0; r < 8; ++r) {
      const int m = r + 8 * sel;
      const float val = o[c][r] / rsumv[m];
      ctx[(size_t)(b * SS + qbase + m) * DM + h * DK + c * 16 + ln] =
          (bf16_t)val;
    }
  }
}

// ------------------------------ launcher -----------------------------------
extern "C" void kernel_launch(void* const* d_in, const int* in_sizes, int n_in,
                              void* d_out, int out_size, void* d_ws,
                              size_t ws_size, hipStream_t stream) {
  const float* q    = (const float*)d_in[0];
  const float* k    = (const float*)d_in[1];
  const float* v    = (const float*)d_in[2];
  const int*   mask = (const int*)d_in[3];
  const float* wq = (const float*)d_in[4];  const float* bq = (const float*)d_in[5];
  const float* wk = (const float*)d_in[6];  const float* bk = (const float*)d_in[7];
  const float* wv = (const float*)d_in[8];  const float* bv = (const float*)d_in[9];
  const float* wo = (const float*)d_in[10]; const float* bo = (const float*)d_in[11];

  char* ws = (char*)d_ws;
  size_t off = 0;
  auto carve = [&](size_t bytes) -> void* {
    void* p = ws + off;
    off += (bytes + 255) & ~(size_t)255;
    return p;
  };

  const size_t ACT  = (size_t)MROWS * DM;     // 4096*1024
  const size_t WSZ  = (size_t)DM * DM;        // 1024*1024
  const size_t HEAD = (size_t)BB * HH * SS * DK;

  bf16_t* Xq = (bf16_t*)carve(ACT * 2);       // later reused as ctx
  bf16_t* Xk = (bf16_t*)carve(ACT * 2);
  bf16_t* Xv = (bf16_t*)carve(ACT * 2);
  bf16_t* Wq = (bf16_t*)carve(WSZ * 2);
  bf16_t* Wk = (bf16_t*)carve(WSZ * 2);
  bf16_t* Wv = (bf16_t*)carve(WSZ * 2);
  bf16_t* Wo = (bf16_t*)carve(WSZ * 2);
  bf16_t* Qb = (bf16_t*)carve(HEAD * 2);
  bf16_t* Kb = (bf16_t*)carve(HEAD * 2);
  bf16_t* Vt = (bf16_t*)carve(HEAD * 2);
  bf16_t* Ctx = Xq;   // Xq dead after Q projection; reuse for merged ctx

  // f32 -> bf16 conversions
  cvt_kernel<<<1024, 256, 0, stream>>>(q, Xq, (int)ACT);
  cvt_kernel<<<1024, 256, 0, stream>>>(k, Xk, (int)ACT);
  cvt_kernel<<<1024, 256, 0, stream>>>(v, Xv, (int)ACT);
  cvt_kernel<<<512, 256, 0, stream>>>(wq, Wq, (int)WSZ);
  cvt_kernel<<<512, 256, 0, stream>>>(wk, Wk, (int)WSZ);
  cvt_kernel<<<512, 256, 0, stream>>>(wv, Wv, (int)WSZ);
  cvt_kernel<<<512, 256, 0, stream>>>(wo, Wo, (int)WSZ);

  // projections: 4096 16x64 tiles / 4 waves per block -> 1024 blocks
  gemm_kernel<0><<<1024, 128, 0, stream>>>(Xq, Wq, bq, Qb);
  gemm_kernel<0><<<1024, 128, 0, stream>>>(Xk, Wk, bk, Kb);
  gemm_kernel<1><<<1024, 128, 0, stream>>>(Xv, Wv, bv, Vt);

  // attention: B*H*(S/16) = 4096 waves, ~33KB dynamic LDS each
  const size_t shbytes = (size_t)(16 * SST + 48) * sizeof(float);
  attn_kernel<<<4096, 32, shbytes, stream>>>(Qb, Kb, Vt, mask, Ctx);

  // output projection, fp32 + bias straight to d_out
  gemm_kernel<2><<<1024, 128, 0, stream>>>(Ctx, Wo, bo, d_out);
}
